// GABlock_37452114821311
// MI455X (gfx1250) — matrix operand order repaired and checked by reading, not compiled, not run on hardware
//
#include <hip/hip_runtime.h>
#include <hip/hip_bf16.h>

// ---------------------------------------------------------------------------
// Graph-attention block for MI455X (gfx1250, wave32).
// GEMMs: V_WMMA_F32_16X16X32_BF16 (fp32 acc). Block tile 128x128, 8 waves of
// 32x64 => 16 WMMA per K-step (K-step 64, 2 chained K-halves). Tiles staged
// into double-buffered LDS with GLOBAL_LOAD_ASYNC_TO_LDS_B128 (ASYNCcnt) so
// fills overlap math. Q/K/V fused into one GEMM (A read once). Weights are
// pre-transposed to [N,K] bf16 so all tile traffic is b128 vectors.
// Edge softmax/aggregation: memory-bound scatter with float atomics.
// ---------------------------------------------------------------------------

#define DD   256   // model dim
#define HHN  8     // heads
#define DKK  32    // head dim (DK == DV)
#define HUSZ 1024  // FFN hidden
#define QKVW 768   // fused q|k|v width

typedef __attribute__((ext_vector_type(16))) __bf16        v16bf;
typedef __attribute__((ext_vector_type(8)))  float         v8f;
typedef __attribute__((ext_vector_type(4)))  unsigned int  v4u;

struct U4x2 { v4u a, b; };

__device__ __forceinline__ unsigned short f2bf(float f) {
  unsigned int u = __float_as_uint(f);
  u += 0x7FFFu + ((u >> 16) & 1u);       // round-to-nearest-even
  return (unsigned short)(u >> 16);
}

__device__ __forceinline__ v16bf make_frag(v4u x, v4u y) {
  U4x2 p{x, y};
  return __builtin_bit_cast(v16bf, p);
}

// float atomic-max via integer ordering trick (works for mixed signs)
__device__ __forceinline__ void atomicMaxF(float* addr, float val) {
  if (val >= 0.0f) atomicMax((int*)addr, __float_as_int(val));
  else             atomicMin((unsigned int*)addr, __float_as_uint(val));
}

// gfx1250 async global->LDS copy (16B per lane), tracked by ASYNCcnt.
__device__ __forceinline__ void async_load_b128(const void* gaddr, void* lds_ptr) {
  unsigned lds_off = (unsigned)(unsigned long long)lds_ptr;  // low 32b = LDS offset
  asm volatile("global_load_async_to_lds_b128 %0, %1, off"
               :: "v"(lds_off), "v"(gaddr) : "memory");
}
__device__ __forceinline__ void wait_async0() {
  asm volatile("s_wait_asynccnt 0" ::: "memory");
}

// ---------------------------------------------------------------------------
// C[M,Ncols] = A[M,K] @ B[K,Ncols], B supplied TRANSPOSED as Bt[Ncols,K]
// (bf16 row-major). Optional bias + relu; fp32 and/or bf16 out.
// Grid: x = Ncols/128, y = ceil(M/128). Block = 256 threads = 8 waves.
// Wave w: rows (w&3)*32..+31, cols (w>>2)*64..+63 of the block tile.
// K stepped by 64 (two chained 16x16x32 WMMAs per output tile per step).
// Out-of-range A rows are CLAMPED to M-1 (C rows >= M are never stored and
// each C row depends only on its own A row), so the loop has no predication.
// Requires: K % 64 == 0, Ncols % 128 == 0.
// ---------------------------------------------------------------------------
__global__ void __launch_bounds__(256)
gemm_bf16_wmma(const unsigned short* __restrict__ A,
               const unsigned short* __restrict__ Bt,
               float* __restrict__ Cf, unsigned short* __restrict__ Cb,
               const float* __restrict__ bias,
               int M, int Ncols, int K, int relu)
{
  // [buf][ A: 128x64 (8192) | Bt: 128x64 (8192) ]  = 64 KB total
  __shared__ unsigned short lds[2][16384];

  const int tid  = threadIdx.x;
  const int wave = tid >> 5;
  const int lane = tid & 31;
  const int ll   = lane & 15;
  const int hi   = lane >> 4;
  const int wm   = wave & 3;       // M subtile (32 rows)
  const int wn   = wave >> 2;      // N subtile (64 cols)
  const int n0   = blockIdx.x * 128;
  const int mBase = blockIdx.y * 128;

  // tile-fill coordinates: 8 x b128 per thread per K-step (4 A rows + 4 B rows)
  const int frow = tid >> 3;       // 0..31
  const int fcq  = tid & 7;        // 16B chunk within a 64-elem row
  const unsigned short* gAr[4];
  const unsigned short* gBr[4];
  int lA[4], lB[4];
  #pragma unroll
  for (int i = 0; i < 4; ++i) {
    int r  = frow + i * 32;
    int ar = min(mBase + r, M - 1);
    gAr[i] = A  + (size_t)ar * K + fcq * 8;
    gBr[i] = Bt + (size_t)(n0 + r) * K + fcq * 8;
    lA[i]  = r * 64 + fcq * 8;
    lB[i]  = 8192 + lA[i];
  }

  v8f acc[2][4];
  #pragma unroll
  for (int i = 0; i < 2; ++i)
    #pragma unroll
    for (int j = 0; j < 4; ++j)
      acc[i][j] = (v8f){0.f, 0.f, 0.f, 0.f, 0.f, 0.f, 0.f, 0.f};

  // prime buffer 0
  #pragma unroll
  for (int i = 0; i < 4; ++i) {
    async_load_b128(gAr[i], &lds[0][lA[i]]);
    async_load_b128(gBr[i], &lds[0][lB[i]]);
  }
  wait_async0();
  __syncthreads();

  int cur = 0;
  for (int k0 = 0; k0 < K; k0 += 64) {
    // prefetch next K-slab into the other buffer (overlaps the WMMAs below)
    if (k0 + 64 < K) {
      int nxt = cur ^ 1;
      #pragma unroll
      for (int i = 0; i < 4; ++i) {
        async_load_b128(gAr[i] + k0 + 64, &lds[nxt][lA[i]]);
        async_load_b128(gBr[i] + k0 + 64, &lds[nxt][lB[i]]);
      }
    }

    const unsigned short* La = &lds[cur][0];
    const unsigned short* Lb = &lds[cur][8192];

    #pragma unroll
    for (int kk = 0; kk < 2; ++kk) {   // two 32-wide K halves
      v16bf af[2], bf[4];
      #pragma unroll
      for (int mi = 0; mi < 2; ++mi) {
        const unsigned short* ar = La + (wm * 32 + mi * 16 + ll) * 64 + kk * 32;
        af[mi] = make_frag(*(const v4u*)(ar + hi * 16),
                           *(const v4u*)(ar + hi * 16 + 8));
      }
      #pragma unroll
      for (int ni = 0; ni < 4; ++ni) {
        const unsigned short* br =
            Lb + (wn * 64 + ni * 16 + ll) * 64 + kk * 32 + hi * 16;
        bf[ni] = make_frag(*(const v4u*)(br), *(const v4u*)(br + 8));
      }
      #pragma unroll
      for (int mi = 0; mi < 2; ++mi)
        #pragma unroll
        for (int ni = 0; ni < 4; ++ni)
          acc[mi][ni] = __builtin_amdgcn_wmma_f32_16x16x32_bf16(
              false, af[mi], false, bf[ni], (short)0, acc[mi][ni], false, false);
    }

    wait_async0();      // our async fills for the next buffer are done
    __syncthreads();    // everyone done reading cur + everyone's fills visible
    cur ^= 1;
  }

  // writeback: VGPR j -> M = j (lanes 0-15) / 8+j (lanes 16-31)
  #pragma unroll
  for (int mi = 0; mi < 2; ++mi) {
    int rbase = mBase + wm * 32 + mi * 16 + (hi << 3);
    #pragma unroll
    for (int ni = 0; ni < 4; ++ni) {
      int col = n0 + wn * 64 + ni * 16 + ll;
      float bval = bias ? bias[col] : 0.f;
      #pragma unroll
      for (int j = 0; j < 8; ++j) {
        int grow = rbase + j;
        if (grow < M) {
          float v = acc[mi][ni][j] + bval;
          if (relu) v = fmaxf(v, 0.f);
          size_t o = (size_t)grow * Ncols + col;
          if (Cf) Cf[o] = v;
          if (Cb) Cb[o] = f2bf(v);
        }
      }
    }
  }
}

// ---------------------------------------------------------------------------
// Elementwise helpers
// ---------------------------------------------------------------------------
__global__ void cvt_f32_bf16(const float* __restrict__ in,
                             unsigned short* __restrict__ out, long n) {
  long i = (long)blockIdx.x * blockDim.x + threadIdx.x;
  long s = (long)gridDim.x * blockDim.x;
  for (long t = i; t < n; t += s) out[t] = f2bf(in[t]);
}

// in: [K,N] fp32 row-major; out: [N,K] bf16 row-major (transposed)
__global__ void cvt_transpose_bf16(const float* __restrict__ in,
                                   unsigned short* __restrict__ out,
                                   int K, int N) {
  long t = (long)blockIdx.x * blockDim.x + threadIdx.x;
  long total = (long)K * N;
  if (t >= total) return;
  int n = (int)(t / K);
  int k = (int)(t % K);
  out[t] = f2bf(in[(size_t)k * N + n]);
}

__global__ void copy_cvt(const float* __restrict__ in,
                         float* __restrict__ outf,
                         unsigned short* __restrict__ outb, long n) {
  long i = (long)blockIdx.x * blockDim.x + threadIdx.x;
  long s = (long)gridDim.x * blockDim.x;
  for (long t = i; t < n; t += s) { float v = in[t]; outf[t] = v; outb[t] = f2bf(v); }
}

__global__ void init_attn(float* __restrict__ m, float* __restrict__ z,
                          float* __restrict__ agg, long nm, long nagg) {
  long i = (long)blockIdx.x * blockDim.x + threadIdx.x;
  long s = (long)gridDim.x * blockDim.x;
  for (long t = i; t < nagg; t += s) {
    agg[t] = 0.f;
    if (t < nm) { m[t] = -3.402823466e38f; z[t] = 0.f; }
  }
}

// ---------------------------------------------------------------------------
// Edge stage (memory/atomic bound): one thread per (edge, head).
// qkv layout: [N, 768] = [q(256) | k(256) | v(256)] per node.
// ---------------------------------------------------------------------------
__global__ void edge_logits(const float* __restrict__ qkv,
                            const int* __restrict__ src, const int* __restrict__ dst,
                            float* __restrict__ e, float* __restrict__ m, int E) {
  int t = blockIdx.x * blockDim.x + threadIdx.x;
  if (t >= E * HHN) return;
  int eid = t >> 3, h = t & 7;
  int s = src[eid], d = dst[eid];
  const float4* qp = (const float4*)(qkv + (size_t)d * QKVW + h * DKK);          // q[dst]
  const float4* kp = (const float4*)(qkv + (size_t)s * QKVW + 256 + h * DKK);    // k[src]
  float acc = 0.f;
  #pragma unroll
  for (int i = 0; i < 8; ++i) {
    float4 a = qp[i], b = kp[i];
    acc += a.x * b.x + a.y * b.y + a.z * b.z + a.w * b.w;
  }
  acc *= 0.17677669529663687f;  // 1/sqrt(32)
  e[t] = acc;
  atomicMaxF(&m[(size_t)d * HHN + h], acc);
}

__global__ void edge_exp(float* __restrict__ e, const int* __restrict__ dst,
                         const float* __restrict__ m, float* __restrict__ z, int E) {
  int t = blockIdx.x * blockDim.x + threadIdx.x;
  if (t >= E * HHN) return;
  int eid = t >> 3, h = t & 7;
  int d = dst[eid];
  float a = __expf(e[t] - m[(size_t)d * HHN + h]);
  e[t] = a;
  atomicAdd(&z[(size_t)d * HHN + h], a);
}

__global__ void edge_agg(const float* __restrict__ e, const float* __restrict__ z,
                         const float* __restrict__ qkv, const int* __restrict__ src,
                         const int* __restrict__ dst, float* __restrict__ agg, int E) {
  int t = blockIdx.x * blockDim.x + threadIdx.x;
  if (t >= E * HHN) return;
  int eid = t >> 3, h = t & 7;
  int s = src[eid], d = dst[eid];
  float alpha = e[t] / (z[(size_t)d * HHN + h] + 1e-9f);
  const float4* vp = (const float4*)(qkv + (size_t)s * QKVW + 512 + h * DKK);    // v[src]
  float* ap = agg + (size_t)d * DD + h * DKK;
  #pragma unroll
  for (int i = 0; i < 8; ++i) {
    float4 tv = vp[i];
    atomicAdd(ap + i * 4 + 0, tv.x * alpha);
    atomicAdd(ap + i * 4 + 1, tv.y * alpha);
    atomicAdd(ap + i * 4 + 2, tv.z * alpha);
    atomicAdd(ap + i * 4 + 3, tv.w * alpha);
  }
}

// ---------------------------------------------------------------------------
// y = LayerNorm(x + resid) * g + b ; one wave32 per node, 8 elems/lane.
// Writes fp32 (residual chain) and bf16 (next GEMM input).
// ---------------------------------------------------------------------------
__global__ void add_layernorm(const float* __restrict__ x, const float* __restrict__ resid,
                              const float* __restrict__ g, const float* __restrict__ b,
                              float* __restrict__ outf, unsigned short* __restrict__ outb,
                              int Nn) {
  int node = (int)((blockIdx.x * (long)blockDim.x + threadIdx.x) >> 5);
  int lane = threadIdx.x & 31;
  if (node >= Nn) return;

  const float* xr = x + (size_t)node * DD;
  const float* rr = resid + (size_t)node * DD;
  float vals[8];
  float sum = 0.f;
  #pragma unroll
  for (int i = 0; i < 8; ++i) {
    float t = xr[lane + i * 32] + rr[lane + i * 32];
    vals[i] = t; sum += t;
  }
  #pragma unroll
  for (int o = 16; o > 0; o >>= 1) sum += __shfl_xor(sum, o, 32);
  float mu = sum * (1.f / 256.f);
  float var = 0.f;
  #pragma unroll
  for (int i = 0; i < 8; ++i) { float dd = vals[i] - mu; var += dd * dd; }
  #pragma unroll
  for (int o = 16; o > 0; o >>= 1) var += __shfl_xor(var, o, 32);
  float inv = rsqrtf(var * (1.f / 256.f) + 1e-5f);
  #pragma unroll
  for (int i = 0; i < 8; ++i) {
    int c = lane + i * 32;
    float y = (vals[i] - mu) * inv * g[c] + b[c];
    outf[(size_t)node * DD + c] = y;
    outb[(size_t)node * DD + c] = f2bf(y);
  }
}

// ---------------------------------------------------------------------------
// Host-side orchestration
// ---------------------------------------------------------------------------
extern "C" void kernel_launch(void* const* d_in, const int* in_sizes, int n_in,
                              void* d_out, int out_size, void* d_ws, size_t ws_size,
                              hipStream_t stream) {
  const float* h    = (const float*)d_in[0];
  const int*   src  = (const int*)d_in[1];
  const int*   dst  = (const int*)d_in[2];
  const float* Wq   = (const float*)d_in[3];
  const float* Wk   = (const float*)d_in[4];
  const float* Wv   = (const float*)d_in[5];
  const float* Wo   = (const float*)d_in[6];
  const float* ln_g = (const float*)d_in[7];
  const float* ln_b = (const float*)d_in[8];
  const float* w1   = (const float*)d_in[9];
  const float* b1   = (const float*)d_in[10];
  const float* w2   = (const float*)d_in[11];
  const float* b2   = (const float*)d_in[12];

  const int N = in_sizes[0] / DD;
  const int E = in_sizes[1];
  const int L = in_sizes[7] / DD;
  const long ND  = (long)N * DD;
  const long EH  = (long)E * HHN;
  const long NH  = (long)N * HHN;

  // workspace carve-out (256B aligned)
  char* p = (char*)d_ws;
  auto carve = [&](size_t bytes) -> char* {
    char* r = p; p += (bytes + 255) & ~size_t(255); return r;
  };
  float*          hcur  = (float*)carve(ND * 4);
  unsigned short* hb    = (unsigned short*)carve(ND * 2);
  float*          qkv   = (float*)carve((long)N * QKVW * 4);
  float*          eb    = (float*)carve(EH * 4);
  float*          mb    = (float*)carve(NH * 4);
  float*          zb    = (float*)carve(NH * 4);
  float*          agg   = (float*)carve(ND * 4);
  unsigned short* aggb  = (unsigned short*)carve(ND * 2);
  float*          tmp   = (float*)carve(ND * 4);
  unsigned short* hidb  = (unsigned short*)carve((long)N * HUSZ * 2);
  unsigned short* wqkvb = (unsigned short*)carve((size_t)QKVW * DD * 2);  // [768,256]
  unsigned short* wob   = (unsigned short*)carve((size_t)DD * 256 * 2);   // [256,256]
  unsigned short* w1b   = (unsigned short*)carve((size_t)DD * HUSZ * 2);  // [1024,256]
  unsigned short* w2b   = (unsigned short*)carve((size_t)HUSZ * DD * 2);  // [256,1024]

  const int TB = 256;
  const long WEl = (long)DD * 256;  // per-projection weight elements

  // h -> hcur (fp32) + hb (bf16)
  copy_cvt<<<dim3((ND + TB - 1) / TB), TB, 0, stream>>>(h, hcur, hb, ND);

  dim3 qkvGrid(QKVW / 128, (N + 127) / 128);
  dim3 woGrid(DD / 128, (N + 127) / 128);
  dim3 edgeGrid((EH + TB - 1) / TB);
  dim3 wGrid((WEl + TB - 1) / TB);

  for (int l = 0; l < L; ++l) {
    // weights -> bf16 transposed [N,K]; q|k|v packed into one [768,256] buffer
    cvt_transpose_bf16<<<wGrid, TB, 0, stream>>>(Wq + (long)l * WEl, wqkvb,            DD, 256);
    cvt_transpose_bf16<<<wGrid, TB, 0, stream>>>(Wk + (long)l * WEl, wqkvb + 256 * DD, DD, 256);
    cvt_transpose_bf16<<<wGrid, TB, 0, stream>>>(Wv + (long)l * WEl, wqkvb + 512 * DD, DD, 256);
    cvt_transpose_bf16<<<wGrid, TB, 0, stream>>>(Wo + (long)l * WEl, wob,              DD, 256);

    // fused Q|K|V projection (one pass over A)
    gemm_bf16_wmma<<<qkvGrid, TB, 0, stream>>>(hb, wqkvb, qkv, nullptr, nullptr, N, QKVW, DD, 0);

    // edge softmax + aggregation
    init_attn<<<dim3((ND + TB - 1) / TB), TB, 0, stream>>>(mb, zb, agg, NH, ND);
    edge_logits<<<edgeGrid, TB, 0, stream>>>(qkv, src, dst, eb, mb, E);
    edge_exp<<<edgeGrid, TB, 0, stream>>>(eb, dst, mb, zb, E);
    edge_agg<<<edgeGrid, TB, 0, stream>>>(eb, zb, qkv, src, dst, agg, E);

    // output projection + residual + LN
    cvt_f32_bf16<<<dim3((ND + TB - 1) / TB), TB, 0, stream>>>(agg, aggb, ND);
    gemm_bf16_wmma<<<woGrid, TB, 0, stream>>>(aggb, wob, tmp, nullptr, nullptr, N, 256, DD, 0);
    add_layernorm<<<dim3((N * 32 + TB - 1) / TB), TB, 0, stream>>>(
        tmp, hcur, ln_g + (long)l * DD, ln_b + (long)l * DD, hcur, hb, N);
  }

  // FFN: relu(h @ w1 + b1) @ w2 + b2  -> d_out (fp32)
  long w1El = (long)DD * HUSZ;
  cvt_transpose_bf16<<<dim3((w1El + TB - 1) / TB), TB, 0, stream>>>(w1, w1b, DD, HUSZ);
  cvt_transpose_bf16<<<dim3((w1El + TB - 1) / TB), TB, 0, stream>>>(w2, w2b, HUSZ, DD);

  dim3 ffn1Grid(HUSZ / 128, (N + 127) / 128);
  gemm_bf16_wmma<<<ffn1Grid, TB, 0, stream>>>(hb, w1b, nullptr, hidb, b1, N, HUSZ, DD, 1);
  dim3 ffn2Grid(DD / 128, (N + 127) / 128);
  gemm_bf16_wmma<<<ffn2Grid, TB, 0, stream>>>(hidb, w2b, (float*)d_out, nullptr, b2, N, DD, HUSZ, 0);
}
